// CTC_32830730010922
// MI455X (gfx1250) — matrix operand order repaired and checked by reading, not compiled, not run on hardware
//
#include <hip/hip_runtime.h>

// Problem sizes (fixed by the reference's setup_inputs)
#define BB 16
#define TT 500
#define DD 256
#define VV 5000
#define LL 50
#define SS (2 * LL + 1)      // 101
#define ROWS (BB * TT)       // 8000
#define NB ((VV + 63) / 64)  // 79 column-blocks of 64
#define NCT (NB * 4)         // 316 padded col-tiles of 16 (>= ceil(V/16)=313)
#define CH (DD / 32)         // 8 K-chunks of 32
#define MT (ROWS / 16)       // 500 row-tiles of 16
#define NEGINF (-1e30f)

typedef __attribute__((ext_vector_type(16))) __bf16 v16bf;
typedef __attribute__((ext_vector_type(8)))  float  v8f;
typedef __attribute__((ext_vector_type(4)))  int    v4i;

// ---- CDNA5 async global->LDS path (guarded so compilation never breaks) ----
#if defined(__HIP_DEVICE_COMPILE__)
#  if __has_builtin(__builtin_amdgcn_global_load_async_to_lds_b128)
#    define HAVE_ASYNC 1
#  else
#    define HAVE_ASYNC 0
#  endif
#  if __has_builtin(__builtin_amdgcn_s_wait_asynccnt)
#    define HAVE_WAITA 1
#  else
#    define HAVE_WAITA 0
#  endif
#else
#  define HAVE_ASYNC 0
#  define HAVE_WAITA 0
#endif

#if HAVE_ASYNC
typedef __attribute__((address_space(1))) v4i g_v4i;  // global ("__device__")
typedef __attribute__((address_space(3))) v4i l_v4i;  // LDS ("__shared__")
#endif

__device__ __forceinline__ void cp_async16(void* lds, const void* g) {
#if HAVE_ASYNC
  __builtin_amdgcn_global_load_async_to_lds_b128(
      (g_v4i*)(void*)g, (l_v4i*)lds, 0, 0);
#else
  *(float4*)lds = *(const float4*)g;
#endif
}
__device__ __forceinline__ void async_wait0() {
#if HAVE_ASYNC
#  if HAVE_WAITA
  __builtin_amdgcn_s_wait_asynccnt(0);
#  else
  asm volatile("s_wait_asynccnt 0x0" ::: "memory");
#  endif
#endif
}

__device__ __forceinline__ float red_max16(float v) {
  #pragma unroll
  for (int m = 8; m; m >>= 1) v = fmaxf(v, __shfl_xor(v, m, 16));
  return v;
}
__device__ __forceinline__ float red_sum16(float v) {
  #pragma unroll
  for (int m = 8; m; m >>= 1) v += __shfl_xor(v, m, 16);
  return v;
}
__device__ __forceinline__ float lae(float a, float b) {
  float mx = fmaxf(a, b), mn = fminf(a, b);
  return mx + log1pf(expf(mn - mx));
}

// ---------------------------------------------------------------------------
// Pack kernels: one-time f32 -> (bf16 hi, bf16 lo) split, written in the
// exact WMMA per-lane fragment order so the GEMM inner loop is load+WMMA only.
// A fragment (16-bit A 16x32): lane l -> row M=(l&15); halves 0..7 = K 8h+j,
// halves 8..15 = K 16+8h+j (h = l>>4). One thread per (tile,chunk,lane).
// ---------------------------------------------------------------------------
__global__ void k_pack_A(const float* __restrict__ A,
                         unsigned short* __restrict__ hi,
                         unsigned short* __restrict__ lo)
{
  const int idx = blockIdx.x * blockDim.x + threadIdx.x;  // tile*256 + chunk*32 + lane
  if (idx >= MT * CH * 32) return;
  const int tile  = idx >> 8;
  const int chunk = (idx >> 5) & 7;
  const int lane  = idx & 31;
  const int half  = lane >> 4, m = lane & 15;
  const float* src = A + (size_t)(tile * 16 + m) * DD + chunk * 32;
  __bf16* oh = (__bf16*)hi + (size_t)idx * 16;
  __bf16* ol = (__bf16*)lo + (size_t)idx * 16;
  #pragma unroll
  for (int j = 0; j < 16; ++j) {
    const int K = (j < 8) ? (8 * half + j) : (16 + 8 * half + (j - 8));
    float v = src[K];
    __bf16 h = (__bf16)v;
    oh[j] = h;
    ol[j] = (__bf16)(v - (float)h);
  }
}

// B fragment (16-bit B 32x16): lane l -> col N=(l&15); halves 0..7 = K 8h+j,
// halves 8..15 = K 16+8h+j. Col-tiles padded to NCT with zeros (branch-free GEMM).
__global__ void k_pack_W(const float* __restrict__ Wm,
                         unsigned short* __restrict__ hi,
                         unsigned short* __restrict__ lo)
{
  const int idx = blockIdx.x * blockDim.x + threadIdx.x;  // ctile*256 + chunk*32 + lane
  if (idx >= NCT * CH * 32) return;
  const int ct    = idx >> 8;
  const int chunk = (idx >> 5) & 7;
  const int lane  = idx & 31;
  const int half  = lane >> 4;
  const int n     = ct * 16 + (lane & 15);
  __bf16* oh = (__bf16*)hi + (size_t)idx * 16;
  __bf16* ol = (__bf16*)lo + (size_t)idx * 16;
  if (n < VV) {
    const float* src = Wm + (size_t)n * DD + chunk * 32;
    #pragma unroll
    for (int j = 0; j < 16; ++j) {
      float v = src[(j < 8) ? (8 * half + j) : (16 + 8 * half + (j - 8))];
      __bf16 h = (__bf16)v;
      oh[j] = h;
      ol[j] = (__bf16)(v - (float)h);
    }
  } else {
    #pragma unroll
    for (int j = 0; j < 16; ++j) { oh[j] = (__bf16)0.0f; ol[j] = (__bf16)0.0f; }
  }
}

// ---------------------------------------------------------------------------
// Kernel 1: GEMM (fp32-accurate via 3 bf16 WMMAs) fused with per-tile softmax
// partial reduction. Grid (MT, NB), 128 threads (4 waves x 16-col subtiles).
// A fragments for the whole row-tile are staged into LDS with the CDNA5
// async-DMA path and shared by all 4 waves; W fragments stream from global
// (L2-resident, fully coalesced 1KB/wave bursts).
// ---------------------------------------------------------------------------
__global__ __launch_bounds__(128)
void k_gemm_lse(const unsigned short* __restrict__ afhi,
                const unsigned short* __restrict__ aflo,
                const unsigned short* __restrict__ wfhi,
                const unsigned short* __restrict__ wflo,
                const float* __restrict__ bias,
                float* __restrict__ pmax, float* __restrict__ psum)
{
  __shared__ __align__(32) unsigned short AhiS[CH * 32 * 16];  // 8 KB
  __shared__ __align__(32) unsigned short AloS[CH * 32 * 16];  // 8 KB
  __shared__ float sM[4][16];
  __shared__ float sS[4][16];

  const int tile  = blockIdx.x;
  const int nb    = blockIdx.y;
  const int wave  = threadIdx.x >> 5;
  const int lane  = threadIdx.x & 31;
  const int l15   = lane & 15;
  const int half  = lane >> 4;
  const int ctile = nb * 4 + wave;           // < NCT (padded)
  const int nCol  = ctile * 16 + l15;
  const bool colOK = (nCol < VV);

  // ---- stage this row-tile's A fragments (hi+lo, 16 KB) into LDS ----
  {
    const unsigned short* gh = afhi + (size_t)tile * (CH * 32 * 16);
    const unsigned short* gl = aflo + (size_t)tile * (CH * 32 * 16);
    for (int i = threadIdx.x; i < CH * 32 * 16 / 8; i += 128) {  // 16B units
      cp_async16(&AhiS[i * 8], gh + i * 8);
      cp_async16(&AloS[i * 8], gl + i * 8);
    }
    async_wait0();
  }
  __syncthreads();

  v8f acc = {};
  const unsigned short* wb_h = wfhi + ((size_t)ctile * CH * 32 + lane) * 16;
  const unsigned short* wb_l = wflo + ((size_t)ctile * CH * 32 + lane) * 16;

  #pragma unroll
  for (int c = 0; c < CH; ++c) {
    v16bf ahi = *(const v16bf*)&AhiS[(c * 32 + lane) * 16];
    v16bf alo = *(const v16bf*)&AloS[(c * 32 + lane) * 16];
    v16bf bhi = *(const v16bf*)(wb_h + (size_t)c * 32 * 16);
    v16bf blo = *(const v16bf*)(wb_l + (size_t)c * 32 * 16);
    acc = __builtin_amdgcn_wmma_f32_16x16x32_bf16(false, ahi, false, bhi,
                                                  (short)0, acc, false, false);
    acc = __builtin_amdgcn_wmma_f32_16x16x32_bf16(false, ahi, false, blo,
                                                  (short)0, acc, false, false);
    acc = __builtin_amdgcn_wmma_f32_16x16x32_bf16(false, alo, false, bhi,
                                                  (short)0, acc, false, false);
  }

  // ---- epilogue: bias, column mask, per-row (max, sum-exp) partials ----
  const float bv = colOK ? bias[nCol] : 0.0f;
  float vals[8];
  #pragma unroll
  for (int r = 0; r < 8; ++r)
    vals[r] = colOK ? (acc[r] + bv) : NEGINF;

  float rmax[8], rsum[8];
  #pragma unroll
  for (int r = 0; r < 8; ++r) {
    rmax[r] = red_max16(vals[r]);
    rsum[r] = red_sum16(expf(vals[r] - rmax[r]));  // masked cols -> exp(-inf)=0
  }

  if (l15 == 0) {
    #pragma unroll
    for (int r = 0; r < 8; ++r) {
      sM[wave][r + 8 * half] = rmax[r];
      sS[wave][r + 8 * half] = rsum[r];
    }
  }
  __syncthreads();

  if (threadIdx.x < 16) {
    const int row = threadIdx.x;
    float M = sM[0][row];
    #pragma unroll
    for (int w = 1; w < 4; ++w) M = fmaxf(M, sM[w][row]);
    float Ssum = 0.0f;
    #pragma unroll
    for (int w = 0; w < 4; ++w) Ssum += sS[w][row] * expf(sM[w][row] - M);
    const size_t idx = (size_t)(tile * 16 + row) * NB + nb;
    pmax[idx] = M;
    psum[idx] = Ssum;
  }
}

// ---------------------------------------------------------------------------
// Kernel 2: combine NB partials per row -> LSE[row], row = b*T + t
// ---------------------------------------------------------------------------
__global__ void k_lse_combine(const float* __restrict__ pmax,
                              const float* __restrict__ psum,
                              float* __restrict__ lse)
{
  const int row = blockIdx.x * blockDim.x + threadIdx.x;
  if (row >= ROWS) return;
  const size_t base = (size_t)row * NB;
  float M = pmax[base];
  for (int i = 1; i < NB; ++i) M = fmaxf(M, pmax[base + i]);
  float S = 0.0f;
  for (int i = 0; i < NB; ++i) S += psum[base + i] * expf(pmax[base + i] - M);
  lse[row] = M + logf(S);
}

// ---------------------------------------------------------------------------
// Kernel 3: lp_ext[t][b][s] = enc[b,t,:]·W[ext(b,s),:] + bias[ext] - LSE[b,t]
// ---------------------------------------------------------------------------
__global__ __launch_bounds__(128)
void k_label_logits(const float* __restrict__ A, const float* __restrict__ Wm,
                    const float* __restrict__ bias, const int* __restrict__ y,
                    const float* __restrict__ lse, float* __restrict__ lp_ext)
{
  const int bs = blockIdx.x;
  const int b  = bs / SS;
  const int s  = bs % SS;
  const int t  = blockIdx.y * blockDim.x + threadIdx.x;
  if (t >= TT) return;
  const int v = (s & 1) ? y[b * LL + (s >> 1)] : 0;

  const float4* a4 = (const float4*)(A + ((size_t)b * TT + t) * DD);
  const float4* w4 = (const float4*)(Wm + (size_t)v * DD);
  float acc = 0.0f;
  #pragma unroll 8
  for (int j = 0; j < DD / 4; ++j) {
    float4 a = a4[j], w = w4[j];
    acc = fmaf(a.x, w.x, acc);
    acc = fmaf(a.y, w.y, acc);
    acc = fmaf(a.z, w.z, acc);
    acc = fmaf(a.w, w.w, acc);
  }
  lp_ext[((size_t)t * BB + b) * SS + s] = acc + bias[v] - lse[b * TT + t];
}

// ---------------------------------------------------------------------------
// Kernel 4: CTC alpha scan. One block of 16 waves; wave b owns batch b.
// ---------------------------------------------------------------------------
#define SP (SS + 2)
__global__ __launch_bounds__(512)
void k_ctc_scan(const float* __restrict__ lp_ext,
                const int* __restrict__ enc_lens,
                const int* __restrict__ y, const int* __restrict__ y_lens,
                float* __restrict__ out)
{
  __shared__ float buf0[BB][SP];
  __shared__ float buf1[BB][SP];
  __shared__ unsigned char skipf[BB][SS];
  __shared__ float ends[BB];

  const int b    = threadIdx.x >> 5;
  const int lane = threadIdx.x & 31;

  for (int s = lane; s < SP; s += 32) { buf0[b][s] = NEGINF; buf1[b][s] = NEGINF; }
  for (int s = lane; s < SS; s += 32) {
    unsigned char sk = 0;
    if ((s & 1) && s >= 3)
      sk = (y[b * LL + (s >> 1)] != y[b * LL + (s >> 1) - 1]) ? 1 : 0;
    skipf[b][s] = sk;
  }
  if (lane == 0) ends[b] = 0.0f;
  const int elen = enc_lens[b];
  const int e0   = 2 * y_lens[b];
  __syncthreads();

  const float* lp0 = lp_ext + (size_t)b * SS;   // t=0 slice
  for (int s = lane; s < SS; s += 32)
    buf0[b][2 + s] = (s < 2) ? lp0[s] : NEGINF;
  __syncthreads();
  if (elen == 1 && lane == 0)
    ends[b] = -lae(buf0[b][2 + e0], buf0[b][2 + e0 - 1]);

  for (int t = 1; t < TT; ++t) {
    const float* lpt = lp_ext + ((size_t)t * BB + b) * SS;
    float* cur = (t & 1) ? &buf1[b][0] : &buf0[b][0];
    const float* prv = (t & 1) ? &buf0[b][0] : &buf1[b][0];
    for (int s = lane; s < SS; s += 32) {
      float m = lae(prv[2 + s], prv[1 + s]);
      if (skipf[b][s]) m = lae(m, prv[s]);
      cur[2 + s] = m + lpt[s];
    }
    __syncthreads();
    if (t == elen - 1 && lane == 0)
      ends[b] = -lae(cur[2 + e0], cur[2 + e0 - 1]);
  }
  __syncthreads();

  if (threadIdx.x == 0) {
    float acc = 0.0f;
    #pragma unroll
    for (int i = 0; i < BB; ++i) acc += ends[i];
    out[0] = acc / (float)BB;
  }
}

// ---------------------------------------------------------------------------
extern "C" void kernel_launch(void* const* d_in, const int* in_sizes, int n_in,
                              void* d_out, int out_size, void* d_ws, size_t ws_size,
                              hipStream_t stream)
{
  const float* enc      = (const float*)d_in[0];
  const int*   enc_lens = (const int*)  d_in[1];
  const int*   y        = (const int*)  d_in[2];
  const int*   y_lens   = (const int*)  d_in[3];
  const float* Wm       = (const float*)d_in[4];
  const float* bias     = (const float*)d_in[5];

  // workspace layout (256B-aligned regions, ~21.7 MB total)
  char* p = (char*)d_ws;
  auto take = [&p](size_t bytes) {
    char* r = p;
    p += (bytes + 255) & ~(size_t)255;
    return r;
  };
  float* pmax = (float*)take((size_t)ROWS * NB * 4);
  float* psum = (float*)take((size_t)ROWS * NB * 4);
  float* lse  = (float*)take((size_t)ROWS * 4);
  float* lpx  = (float*)take((size_t)TT * BB * SS * 4);
  unsigned short* afhi = (unsigned short*)take((size_t)MT  * CH * 32 * 16 * 2);
  unsigned short* aflo = (unsigned short*)take((size_t)MT  * CH * 32 * 16 * 2);
  unsigned short* wfhi = (unsigned short*)take((size_t)NCT * CH * 32 * 16 * 2);
  unsigned short* wflo = (unsigned short*)take((size_t)NCT * CH * 32 * 16 * 2);

  k_pack_A<<<(MT * CH * 32 + 255) / 256, 256, 0, stream>>>(enc, afhi, aflo);
  k_pack_W<<<(NCT * CH * 32 + 255) / 256, 256, 0, stream>>>(Wm, wfhi, wflo);

  dim3 g1(MT, NB);
  k_gemm_lse<<<g1, 128, 0, stream>>>(afhi, aflo, wfhi, wflo, bias, pmax, psum);

  k_lse_combine<<<(ROWS + 255) / 256, 256, 0, stream>>>(pmax, psum, lse);

  dim3 g3(BB * SS, (TT + 127) / 128);
  k_label_logits<<<g3, 128, 0, stream>>>(enc, Wm, bias, y, lse, lpx);

  k_ctc_scan<<<1, 512, 0, stream>>>(lpx, enc_lens, y, y_lens, (float*)d_out);
}
